// Encoder_45466523795555
// MI455X (gfx1250) — matrix operand order repaired and checked by reading, not compiled, not run on hardware
//
#include <hip/hip_runtime.h>
#include <cmath>

typedef __attribute__((ext_vector_type(16))) _Float16 v16h;
typedef __attribute__((ext_vector_type(8)))  _Float16 v8h;
typedef __attribute__((ext_vector_type(8)))  float    v8f;

#define Bn 16
#define Ln 256
#define Cn 32
#define Hn 256
#define KHn 128
#define HDECn 256
#define NBn 3
#define KPAD (KHn + 8)   // padded LDS row stride in halves (272B rows, 16B aligned)

// ------------------------------------------------------------------
// Stage 1: one workgroup per (b,c). 256 threads = 8 waves (wave32).
//   Phase A: h1 = relu(t*w1+b1) for both MLPs -> f16 in LDS [256][128]
//   Phase B: per wave: two 16-col h-tiles; per 16-row l-tile do
//            D += A(h1,f16) x B(w2,f16) via v_wmma_f32_16x16x32_f16,
//            fuse X load with online softmax over L and the weighted sum.
// Output z[b,c,h] (+channel_bias, *ch_mask) written into d_out.
// ------------------------------------------------------------------
__global__ __launch_bounds__(256) void stage1_kernel(
    const float* __restrict__ T, const float* __restrict__ X,
    const unsigned char* __restrict__ Mmask,
    const float* __restrict__ ik_w1, const float* __restrict__ ik_b1,
    const float* __restrict__ ik_w2, const float* __restrict__ ik_b2,
    const float* __restrict__ k_w1,  const float* __restrict__ k_b1,
    const float* __restrict__ k_w2,  const float* __restrict__ k_b2,
    const float* __restrict__ channel_bias,
    float* __restrict__ zout)
{
  extern __shared__ unsigned char smem_raw[];
  _Float16* h1ik = (_Float16*)smem_raw;            // [Ln][KPAD] f16
  _Float16* h1k  = h1ik + (size_t)Ln * KPAD;       // [Ln][KPAD] f16
  float* w1b1    = (float*)(h1k + (size_t)Ln * KPAD); // ik_w1,ik_b1,k_w1,k_b1
  float* pen     = w1b1 + 4 * KHn;                 // [Ln] softmax mask penalty
  int*   ivalid  = (int*)(pen + Ln);

  const int tid = threadIdx.x;
  const int b   = blockIdx.x / Cn;
  const int c   = blockIdx.x % Cn;

  if (tid == 0) *ivalid = 0;
  if (tid < KHn) {
    w1b1[0*KHn + tid] = ik_w1[tid];
    w1b1[1*KHn + tid] = ik_b1[tid];
    w1b1[2*KHn + tid] = k_w1[tid];
    w1b1[3*KHn + tid] = k_b1[tid];
  }
  __syncthreads();

  { // ---- Phase A: thread tid fills row l = tid of both h1 arrays ----
    const int l = tid;
    const float tv = T[(size_t)(b*Ln + l)*Cn + c];
    const unsigned char mv = Mmask[(size_t)(b*Ln + l)*Cn + c];
    pen[l] = mv ? 0.0f : -1.0e10f;
    if (mv) atomicOr(ivalid, 1);
    _Float16* rik = h1ik + (size_t)l * KPAD;
    _Float16* rk  = h1k  + (size_t)l * KPAD;
    #pragma unroll 8
    for (int k = 0; k < KHn; ++k) {
      float a = tv * w1b1[0*KHn + k] + w1b1[1*KHn + k];
      float d = tv * w1b1[2*KHn + k] + w1b1[3*KHn + k];
      rik[k] = (_Float16)fmaxf(a, 0.0f);
      rk[k]  = (_Float16)fmaxf(d, 0.0f);
    }
  }
  __syncthreads();

  // ---- Phase B ----
  const float chm = (*ivalid) ? 1.0f : 0.0f;
  const int wv   = tid >> 5;
  const int lane = tid & 31;
  const int nlo  = lane & 15;   // D/B column (N) and A row within tile (M)
  const int hi   = lane >> 4;   // upper-half lane group

  for (int th = 0; th < 2; ++th) {
    const int h0 = (wv*2 + th) * 16;

    // B fragments: lane holds column n=lane&15; halves j -> K = j + 16*hi.
    v16h bik[4], bk[4];
    #pragma unroll
    for (int ks = 0; ks < 4; ++ks) {
      v16h t1, t2;
      #pragma unroll
      for (int j = 0; j < 16; ++j) {
        const int k = ks*32 + j + 16*hi;
        t1[j] = (_Float16)ik_w2[(size_t)k*Hn + h0 + nlo];
        t2[j] = (_Float16)k_w2 [(size_t)k*Hn + h0 + nlo];
      }
      bik[ks] = t1; bk[ks] = t2;
    }
    const float bias_ik = ik_b2[h0 + nlo];
    const float bias_k  = k_b2 [h0 + nlo];

    float m = -3.0e38f, s = 0.0f, zacc = 0.0f;   // online softmax state

    for (int lt = 0; lt < 16; ++lt) {
      const int arow = lt*16 + nlo;
      const _Float16* pikrow = h1ik + (size_t)arow * KPAD;
      const _Float16* pkrow  = h1k  + (size_t)arow * KPAD;
      v8f dik = {}; v8f dk = {};
      #pragma unroll
      for (int ks = 0; ks < 4; ++ks) {
        // A halves j=0..15 -> K=ks*32+16*hi+j
        const int o = ks*32 + 16*hi;
        v8h alo = *(const v8h*)(pikrow + o);
        v8h ahi = *(const v8h*)(pikrow + o + 8);
        v16h aik = __builtin_shufflevector(alo, ahi, 0,1,2,3,4,5,6,7,8,9,10,11,12,13,14,15);
        v8h clo = *(const v8h*)(pkrow + o);
        v8h chi = *(const v8h*)(pkrow + o + 8);
        v16h ak  = __builtin_shufflevector(clo, chi, 0,1,2,3,4,5,6,7,8,9,10,11,12,13,14,15);
        dik = __builtin_amdgcn_wmma_f32_16x16x32_f16(false, aik, false, bik[ks], (short)0, dik, false, false);
        dk  = __builtin_amdgcn_wmma_f32_16x16x32_f16(false, ak,  false, bk[ks],  (short)0, dk,  false, false);
      }
      // D rows for this lane: l = lt*16 + r + 8*hi, column h0+nlo
      const int lbase = lt*16 + 8*hi;
      const float* xp = X + (((size_t)(b*Ln + lbase)*Cn + c)*Hn + h0 + nlo);
      float xv[8], wvv[8];
      float tmax = -3.0e38f;
      #pragma unroll
      for (int r = 0; r < 8; ++r) {
        xv[r] = xp[(size_t)r * Cn * Hn];
        float w = dk[r] + bias_k + xv[r] + pen[lbase + r];
        wvv[r] = w;
        tmax = fmaxf(tmax, w);
      }
      const float nm = fmaxf(m, tmax);
      const float sc = __expf(m - nm);
      s *= sc; zacc *= sc;
      #pragma unroll
      for (int r = 0; r < 8; ++r) {
        float e = __expf(wvv[r] - nm);
        s += e;
        zacc += xv[r] * (dik[r] + bias_ik) * e;
      }
      m = nm;
    }

    // Combine lane and lane^16 (both hold column h0+nlo, different L halves)
    const float m2 = __shfl_xor(m, 16, 32);
    const float s2 = __shfl_xor(s, 16, 32);
    const float z2 = __shfl_xor(zacc, 16, 32);
    const float Mx = fmaxf(m, m2);
    const float S  = s*__expf(m - Mx) + s2*__expf(m2 - Mx);
    const float Zt = zacc*__expf(m - Mx) + z2*__expf(m2 - Mx);
    const float res = (Zt / S) * chm + channel_bias[(size_t)c*Hn + h0 + nlo];
    if (hi == 0)
      zout[((size_t)b*Cn + c)*Hn + h0 + nlo] = res;
  }
}

// ------------------------------------------------------------------
// Stage 2: one workgroup per batch element. z[b] (32x256 fp32) in LDS.
// Runs 3 mixer blocks + init channel mixer + final rmsnorm@out_w.
// Thread tid owns column h = tid across all 32 channels.
// ------------------------------------------------------------------
__device__ inline float block_rms_rs(float ssq, float* red) {
  const int tid = threadIdx.x;
  red[tid] = ssq;
  __syncthreads();
  for (int off = 128; off > 0; off >>= 1) {
    if (tid < off) red[tid] += red[tid + off];
    __syncthreads();
  }
  const float total = red[0];
  __syncthreads();
  return 1.0f / sqrtf(total / (float)(Cn * Hn) + 1.1920928955078125e-07f);
}

__global__ __launch_bounds__(256) void mixer_kernel(
    float* __restrict__ zio,
    const float* __restrict__ cm_rms, const float* __restrict__ cm_w, const float* __restrict__ cm_b,
    const float* __restrict__ km_rms, const float* __restrict__ km_w, const float* __restrict__ km_b,
    const float* __restrict__ icm_rms, const float* __restrict__ icm_w, const float* __restrict__ icm_b,
    const float* __restrict__ out_rms, const float* __restrict__ out_w, const float* __restrict__ out_b)
{
  extern __shared__ unsigned char smem_raw[];
  float* z   = (float*)smem_raw;     // [Cn][Hn]
  float* zc  = z  + Cn*Hn;           // [Cn][Hn]
  float* zn  = zc + Cn*Hn;           // [Cn][Hn] normalized scratch
  float* red = zn + Cn*Hn;           // [256]

  const int tid = threadIdx.x;       // column h
  const int b   = blockIdx.x;

  #pragma unroll
  for (int cc = 0; cc < Cn; ++cc)
    z[cc*Hn + tid] = zio[((size_t)b*Cn + cc)*Hn + tid];
  __syncthreads();

  for (int blk = 0; blk < NBn; ++blk) {
    // ---- channel mix: zc = z + relu(rmsnorm(z^T)@cm_w + cm_b)^T ----
    float ssq = 0.0f;
    #pragma unroll
    for (int cc = 0; cc < Cn; ++cc) { float v = z[cc*Hn + tid]; ssq += v*v; }
    float rs = block_rms_rs(ssq, red);
    #pragma unroll
    for (int cc = 0; cc < Cn; ++cc)
      zn[cc*Hn + tid] = z[cc*Hn + tid] * rs * cm_rms[((size_t)blk*Hn + tid)*Cn + cc];
    __syncthreads();
    float zcreg[Cn];
    #pragma unroll
    for (int co = 0; co < Cn; ++co) {
      float acc = cm_b[blk*Cn + co];
      #pragma unroll
      for (int ci = 0; ci < Cn; ++ci)
        acc += zn[ci*Hn + tid] * cm_w[((size_t)blk*Cn + ci)*Cn + co];
      zcreg[co] = z[co*Hn + tid] + fmaxf(acc, 0.0f);
    }
    __syncthreads();
    #pragma unroll
    for (int cc = 0; cc < Cn; ++cc) zc[cc*Hn + tid] = zcreg[cc];
    __syncthreads();

    // ---- hidden mix: zc2 = zc + relu(rmsnorm(zc)@km_w + km_b); z = z + zc2 ----
    ssq = 0.0f;
    #pragma unroll
    for (int cc = 0; cc < Cn; ++cc) { float v = zc[cc*Hn + tid]; ssq += v*v; }
    float rs2 = block_rms_rs(ssq, red);
    #pragma unroll
    for (int cc = 0; cc < Cn; ++cc)
      zn[cc*Hn + tid] = zc[cc*Hn + tid] * rs2 * km_rms[((size_t)blk*Cn + cc)*Hn + tid];
    __syncthreads();
    float zreg[Cn];
    #pragma unroll
    for (int cc = 0; cc < Cn; ++cc) {
      float acc = km_b[blk*Hn + tid];
      #pragma unroll 8
      for (int h0 = 0; h0 < Hn; ++h0)
        acc += zn[cc*Hn + h0] * km_w[((size_t)blk*Hn + h0)*Hn + tid];
      zreg[cc] = z[cc*Hn + tid] + zc[cc*Hn + tid] + fmaxf(acc, 0.0f);
    }
    __syncthreads();
    #pragma unroll
    for (int cc = 0; cc < Cn; ++cc) z[cc*Hn + tid] = zreg[cc];
    __syncthreads();
  }

  // ---- init channel mixer ----
  {
    float ssq = 0.0f;
    #pragma unroll
    for (int cc = 0; cc < Cn; ++cc) { float v = z[cc*Hn + tid]; ssq += v*v; }
    float rs = block_rms_rs(ssq, red);
    #pragma unroll
    for (int cc = 0; cc < Cn; ++cc)
      zn[cc*Hn + tid] = z[cc*Hn + tid] * rs * icm_rms[(size_t)tid*Cn + cc];
    __syncthreads();
    float zreg[Cn];
    #pragma unroll
    for (int co = 0; co < Cn; ++co) {
      float acc = icm_b[co];
      #pragma unroll
      for (int ci = 0; ci < Cn; ++ci)
        acc += zn[ci*Hn + tid] * icm_w[(size_t)ci*Cn + co];
      zreg[co] = z[co*Hn + tid] + fmaxf(acc, 0.0f);
    }
    __syncthreads();
    #pragma unroll
    for (int cc = 0; cc < Cn; ++cc) z[cc*Hn + tid] = zreg[cc];
    __syncthreads();
  }

  // ---- final: out = rmsnorm(z, out_rms) @ out_w + out_b ----
  {
    float ssq = 0.0f;
    #pragma unroll
    for (int cc = 0; cc < Cn; ++cc) { float v = z[cc*Hn + tid]; ssq += v*v; }
    float rs = block_rms_rs(ssq, red);
    #pragma unroll
    for (int cc = 0; cc < Cn; ++cc)
      zn[cc*Hn + tid] = z[cc*Hn + tid] * rs * out_rms[(size_t)cc*Hn + tid];
    __syncthreads();
    #pragma unroll
    for (int cc = 0; cc < Cn; ++cc) {
      float acc = out_b[tid];
      #pragma unroll 8
      for (int h0 = 0; h0 < Hn; ++h0)
        acc += zn[cc*Hn + h0] * out_w[(size_t)h0*HDECn + tid];
      zio[((size_t)b*Cn + cc)*HDECn + tid] = acc;
    }
  }
}

extern "C" void kernel_launch(void* const* d_in, const int* in_sizes, int n_in,
                              void* d_out, int out_size, void* d_ws, size_t ws_size,
                              hipStream_t stream) {
  (void)in_sizes; (void)n_in; (void)out_size; (void)d_ws; (void)ws_size;
  const float* T            = (const float*)d_in[0];
  const float* X            = (const float*)d_in[1];
  const unsigned char* M    = (const unsigned char*)d_in[2];
  const float* ik_w1        = (const float*)d_in[3];
  const float* ik_b1        = (const float*)d_in[4];
  const float* ik_w2        = (const float*)d_in[5];
  const float* ik_b2        = (const float*)d_in[6];
  const float* k_w1         = (const float*)d_in[7];
  const float* k_b1         = (const float*)d_in[8];
  const float* k_w2         = (const float*)d_in[9];
  const float* k_b2         = (const float*)d_in[10];
  const float* channel_bias = (const float*)d_in[11];
  const float* cm_rms       = (const float*)d_in[12];
  const float* cm_w         = (const float*)d_in[13];
  const float* cm_b         = (const float*)d_in[14];
  const float* km_rms       = (const float*)d_in[15];
  const float* km_w         = (const float*)d_in[16];
  const float* km_b         = (const float*)d_in[17];
  const float* icm_rms      = (const float*)d_in[18];
  const float* icm_w        = (const float*)d_in[19];
  const float* icm_b        = (const float*)d_in[20];
  const float* out_rms      = (const float*)d_in[21];
  const float* out_w        = (const float*)d_in[22];
  const float* out_b        = (const float*)d_in[23];
  float* out = (float*)d_out;

  const size_t sh1 = (size_t)2 * Ln * KPAD * sizeof(_Float16)
                   + (size_t)(4*KHn + Ln) * sizeof(float) + 16;
  const size_t sh2 = (size_t)(3 * Cn * Hn + 256) * sizeof(float);

  // Allow >64KB dynamic LDS (gfx1250 WGP has 320KB).
  (void)hipFuncSetAttribute((const void*)stage1_kernel,
      hipFuncAttributeMaxDynamicSharedMemorySize, (int)sh1);
  (void)hipFuncSetAttribute((const void*)mixer_kernel,
      hipFuncAttributeMaxDynamicSharedMemorySize, (int)sh2);

  stage1_kernel<<<dim3(Bn*Cn), dim3(256), sh1, stream>>>(
      T, X, M, ik_w1, ik_b1, ik_w2, ik_b2, k_w1, k_b1, k_w2, k_b2,
      channel_bias, out);

  mixer_kernel<<<dim3(Bn), dim3(256), sh2, stream>>>(
      out, cm_rms, cm_w, cm_b, km_rms, km_w, km_b,
      icm_rms, icm_w, icm_b, out_rms, out_w, out_b);
}
